// SINCERELoss_84146999263362
// MI455X (gfx1250) — matrix operand order repaired and checked by reading, not compiled, not run on hardware
//
#include <hip/hip_runtime.h>
#include <stdint.h>

// ---------------------------------------------------------------------------
// SINCERE loss for MI455X (gfx1250): bf16 WMMA GEMM over E·E^T, two passes,
// triangular tile schedule (symmetry), double-buffered async global->LDS.
// ---------------------------------------------------------------------------

#define SB   8192          // batch
#define SD   1024          // dim
#define NCLS 100
#define INV_T 14.2857142857142857f   // 1/0.07

typedef __attribute__((ext_vector_type(16))) __bf16 v16bf;
typedef __attribute__((ext_vector_type(8)))  float  v8f;

union Frag {               // one 16x32 (A) or 32x16 (B) bf16 fragment per lane
    v16bf v;
    uint4 q[2];
};

static constexpr int WGM        = 256;  // rows per workgroup panel
static constexpr int WGN        = 128;  // cols per tile step
static constexpr int KSTEP      = 32;   // bf16 WMMA K depth
static constexpr int LDS_STRIDE = 80;   // 64B K-slice + 16B pad (conflict-free b128)
static constexpr int NSLICE     = WGM + WGN;      // 384 row-slices per K step
static constexpr int JCHUNK     = 512;  // columns per workgroup (4 tiles of 128)

// ---- CDNA5 async global->LDS staging (guarded; fallback = load+ds_store) ----
#if __has_builtin(__builtin_amdgcn_global_load_async_to_lds_b128)
#define HAVE_ASYNC_LDS 1
#else
#define HAVE_ASYNC_LDS 0
#endif

// builtin takes int4-vector pointers: (as(1) int4*, as(3) int4*, imm, imm)
typedef int i4 __attribute__((vector_size(16)));
typedef __attribute__((address_space(1))) i4 gi4_t;
typedef __attribute__((address_space(3))) i4 li4_t;

__device__ __forceinline__ gi4_t* as_global4(const void* p) {
    return (gi4_t*)(uintptr_t)p;
}
__device__ __forceinline__ li4_t* as_lds4(void* p) {
    return (li4_t*)(uint32_t)(uintptr_t)p;   // low 32 bits = LDS offset
}
__device__ __forceinline__ void wait_stage() {
#if HAVE_ASYNC_LDS
#if __has_builtin(__builtin_amdgcn_s_wait_asynccnt)
    __builtin_amdgcn_s_wait_asynccnt(0);
#else
    asm volatile("s_wait_asynccnt 0x0" ::: "memory");
#endif
#endif  // fallback path: barrier's dscnt wait covers the ds_stores
}

// ------------------------------ utility kernels ----------------------------

__global__ __launch_bounds__(256) void k_init(float* rowsum, int* hist, float* out) {
    int i = blockIdx.x * 256 + threadIdx.x;
    if (i < SB)   rowsum[i] = 0.0f;
    if (i < NCLS) hist[i]   = 0;
    if (i == 0)   out[0]    = 0.0f;
}

__global__ __launch_bounds__(256) void k_convert(const float* __restrict__ e,
                                                 __bf16* __restrict__ ebf) {
    int i = blockIdx.x * 256 + threadIdx.x;   // over SB*SD
    ebf[i] = (__bf16)e[i];
}

__global__ __launch_bounds__(256) void k_hist(const int* __restrict__ labels, int* hist) {
    int i = blockIdx.x * 256 + threadIdx.x;
    if (i < SB) atomicAdd(&hist[labels[i]], 1);
}

__global__ __launch_bounds__(256) void k_invcnt(const int* __restrict__ labels,
                                                const int* __restrict__ hist,
                                                float* __restrict__ invcnt) {
    int i = blockIdx.x * 256 + threadIdx.x;
    if (i < SB) {
        int c = hist[labels[i]] - 1;          // same-label off-diagonal count
        if (c < 1) c = 1;                     // count_safe guard
        invcnt[i] = 1.0f / (float)c;
    }
}

__global__ __launch_bounds__(256) void k_bd(const float* __restrict__ rowsum,
                                            float* __restrict__ bdv) {
    int i = blockIdx.x * 256 + threadIdx.x;
    if (i < SB) bdv[i] = __logf(rowsum[i]) + INV_T;   // log-denominator per row
}

// ------------------------------ GEMM passes --------------------------------
// Triangular schedule: for row panel R (256 rows) and 128-col tile C:
//   C <  R        : skipped (covered by mirror of the transposed tile)
//   R <= C < R+256: "diagonal square" -> row-side contributions only
//   C >= R+256    : "upper" -> row-side AND mirrored column-side contributions
// PASS==1: rowsum[i] += masked exp((dot-1)/T) row sums (+ column sums if upper)
// PASS==2: out += log1p(exp(bd[i]-logit))*ic[j] (+ mirror bd[j],ic[i] if upper)
template <int PASS>
__global__ __launch_bounds__(256)
void sincere_gemm(const __bf16* __restrict__ E,
                  const int*    __restrict__ labels,
                  const float*  __restrict__ invcnt,
                  float*        __restrict__ rowsum,
                  const float*  __restrict__ bdv,
                  float*        __restrict__ out) {
    __shared__ char  ldsAB[2][NSLICE * LDS_STRIDE];  // A rows [0,256), B [256,384)
    __shared__ int   labR[WGM];
    __shared__ int   labC[WGN];
    __shared__ float bdR[WGM];      // pass2: row-side log-denominators
    __shared__ float icR[WGM];      // pass2: row-side 1/count (mirror term)
    __shared__ float red[8];        // pass2: per-wave partials

    const int rowBase = blockIdx.x * WGM;     // 32 row panels
    const int tid  = threadIdx.x;
    const int lane = tid & 31;
    const int wave = tid >> 5;
    const int wi   = wave >> 1;               // 0..3 -> 64-row strip
    const int wj   = wave & 1;                // 0..1 -> 64-col strip
    const int l15  = lane & 15;
    const int hi   = lane >> 4;

    {   // stage row-side arrays once (256 threads cover WGM=256)
        labR[tid] = labels[rowBase + tid];
        if (PASS == 2) {
            bdR[tid] = bdv[rowBase + tid];
            icR[tid] = invcnt[rowBase + tid];
        }
    }

    float lsum = 0.0f;                        // pass2 local loss partial

    for (int jt = 0; jt < JCHUNK / WGN; ++jt) {
        const int colBase = blockIdx.y * JCHUNK + jt * WGN;
        if (colBase < rowBase) continue;              // lower triangle: skip (uniform)
        const bool upper = colBase >= rowBase + WGM;  // mirrored contributions?

        __syncthreads();                      // protect labC & LDS buffers across jt
        if (tid < WGN) labC[tid] = labels[colBase + tid];

        // stage 384 row-slices x 64B (1536 16B chunks over 256 threads) into buf
        auto stage = [&](char* buf, int kb) {
#pragma unroll
            for (int s = 0; s < 6; ++s) {
                const int c    = tid + 256 * s;      // 0..1535
                const int row  = c >> 2;
                const int part = c & 3;
                const __bf16* g = (row < WGM)
                    ? E + (size_t)(rowBase + row) * SD + kb
                    : E + (size_t)(colBase + (row - WGM)) * SD + kb;
                char* l = buf + row * LDS_STRIDE + part * 16;
#if HAVE_ASYNC_LDS
                __builtin_amdgcn_global_load_async_to_lds_b128(
                    as_global4((const uint4*)g + part), as_lds4(l), 0, 0);
#else
                *(uint4*)l = *((const uint4*)g + part);
#endif
            }
        };

        v8f acc[4][4];
#pragma unroll
        for (int a = 0; a < 4; ++a)
#pragma unroll
            for (int b = 0; b < 4; ++b) acc[a][b] = (v8f)(0.0f);

        stage(ldsAB[0], 0);                   // prologue: fill buffer 0
        for (int kb = 0; kb < SD; kb += KSTEP) {
            const int cur = (kb >> 5) & 1;
            wait_stage();                     // own async writes to cur done
            __syncthreads();                  // everyone's cur writes visible;
                                              // also WAR-clears the other buffer
            if (kb + KSTEP < SD) stage(ldsAB[cur ^ 1], kb + KSTEP);  // overlap

            const char* base0 = ldsAB[cur];
            // A fragments: lane half hi holds K {hi*8..+7, 16+hi*8..+7}
            Frag a[4];
#pragma unroll
            for (int ti = 0; ti < 4; ++ti) {
                const char* base = base0 + (wi * 64 + ti * 16 + l15) * LDS_STRIDE;
                a[ti].q[0] = *(const uint4*)(base + hi * 16);
                a[ti].q[1] = *(const uint4*)(base + 32 + hi * 16);
            }
            // B fragments: lane half hi holds K {hi*8..+7, 16+hi*8..+7}
#pragma unroll
            for (int tj = 0; tj < 4; ++tj) {
                const char* base = base0 +
                    (WGM + wj * 64 + tj * 16 + l15) * LDS_STRIDE;
                Frag b;
                b.q[0] = *(const uint4*)(base + hi * 16);
                b.q[1] = *(const uint4*)(base + 32 + hi * 16);
#pragma unroll
                for (int ti = 0; ti < 4; ++ti)
                    acc[ti][tj] = __builtin_amdgcn_wmma_f32_16x16x32_bf16(
                        false, a[ti].v, false, b.v, (short)0, acc[ti][tj], false, false);
            }
        }

        // ---- epilogue: C/D layout -> lane l15 = N, VGPR r = M (+8 if lane>=16)
        float rs[4][8];                       // pass1: jt-local row partials
        float cs[4];                          // pass1: jt-local column partials
#pragma unroll
        for (int a = 0; a < 4; ++a) {
            cs[a] = 0.0f;
#pragma unroll
            for (int b = 0; b < 8; ++b) rs[a][b] = 0.0f;
        }

#pragma unroll
        for (int ti = 0; ti < 4; ++ti) {
            const int mrow = wi * 64 + ti * 16 + hi * 8;
#pragma unroll
            for (int tj = 0; tj < 4; ++tj) {
                const int ncol = wj * 64 + tj * 16 + l15;
                const int gj   = colBase + ncol;
                const int lj   = labC[ncol];
                const float icc = (PASS == 2) ? invcnt[gj] : 0.0f;  // L1-hot
                const float bdc = (PASS == 2 && upper) ? bdv[gj] : 0.0f;
#pragma unroll
                for (int r = 0; r < 8; ++r) {
                    const float dot = acc[ti][tj][r];
                    const int m  = mrow + r;
                    const int li = labR[m];
                    if (PASS == 1) {
                        // fixed-shift softmax: dot<=1 -> exponent <= 0
                        const float e = (li != lj) ? __expf((dot - 1.0f) * INV_T) : 0.0f;
                        rs[ti][r] += e;
                        if (upper) cs[tj] += e;      // mirrored (j,i) row sum
                    } else {
                        const int gi = rowBase + m;
                        if (li == lj && gi != gj) {
                            const float lg = dot * INV_T;
                            lsum += log1pf(__expf(bdR[m] - lg)) * icc;
                            if (upper)
                                lsum += log1pf(__expf(bdc - lg)) * icR[m];
                        }
                    }
                }
            }
        }

        if (PASS == 1) {
            // row sums: reduce over the 16 lanes of each half, 1 atomic/row
#pragma unroll
            for (int ti = 0; ti < 4; ++ti)
#pragma unroll
                for (int r = 0; r < 8; ++r) {
                    float s = rs[ti][r];
                    s += __shfl_xor(s, 1);
                    s += __shfl_xor(s, 2);
                    s += __shfl_xor(s, 4);
                    s += __shfl_xor(s, 8);
                    if (l15 == 0)
                        atomicAdd(&rowsum[rowBase + wi * 64 + ti * 16 + hi * 8 + r], s);
                }
            if (upper) {
                // column sums (mirror): combine the two lane halves, 1 atomic/col
#pragma unroll
                for (int tj = 0; tj < 4; ++tj) {
                    float s = cs[tj];
                    s += __shfl_xor(s, 16);
                    if (hi == 0)
                        atomicAdd(&rowsum[colBase + wj * 64 + tj * 16 + l15], s);
                }
            }
        }
    }

    if (PASS == 2) {
        for (int m = 1; m < 32; m <<= 1) lsum += __shfl_xor(lsum, m);
        if (lane == 0) red[wave] = lsum;
        __syncthreads();
        if (tid == 0) {
            float t = 0.0f;
#pragma unroll
            for (int w = 0; w < 8; ++w) t += red[w];
            atomicAdd(out, t * (1.0f / (float)SB));
        }
    }
}

// ------------------------------ launcher -----------------------------------

extern "C" void kernel_launch(void* const* d_in, const int* in_sizes, int n_in,
                              void* d_out, int out_size, void* d_ws, size_t ws_size,
                              hipStream_t stream) {
    const float* embeds = (const float*)d_in[0];   // (B, D) fp32, L2-normalized
    const int*   labels = (const int*)d_in[1];     // (B,) int32
    float* out = (float*)d_out;                    // scalar

    // workspace layout (16B-aligned slices)
    char* ws = (char*)d_ws;
    __bf16* ebf   = (__bf16*)ws;                                // 16 MiB
    float*  rowsm = (float*)(ws + (size_t)SB * SD * 2);         // 32 KiB
    float*  icnt  = (float*)(ws + (size_t)SB * SD * 2 + SB * 4);
    float*  bdv   = (float*)(ws + (size_t)SB * SD * 2 + SB * 8);
    int*    hist  = (int*)  (ws + (size_t)SB * SD * 2 + SB * 12);

    k_init   <<<SB / 256, 256, 0, stream>>>(rowsm, hist, out);
    k_convert<<<(SB * SD) / 256, 256, 0, stream>>>(embeds, ebf);
    k_hist   <<<SB / 256, 256, 0, stream>>>(labels, hist);
    k_invcnt <<<SB / 256, 256, 0, stream>>>(labels, hist, icnt);

    dim3 grid(SB / WGM, SB / JCHUNK);   // 32 x 16 workgroups, 256 threads each
    sincere_gemm<1><<<grid, 256, 0, stream>>>(ebf, labels, icnt, rowsm, bdv, out);
    k_bd     <<<SB / 256, 256, 0, stream>>>(rowsm, bdv);
    sincere_gemm<2><<<grid, 256, 0, stream>>>(ebf, labels, icnt, rowsm, bdv, out);
}
